// DualAttention_24206435680953
// MI455X (gfx1250) — compile-verified, matrix operand order checked by reading, not run on hardware
//
#include <hip/hip_runtime.h>

// ---------------------------------------------------------------------------
// DualAttention fused implementation for gfx1250 (MI455X, wave32, WMMA bf16,
// TDM tensor_load_to_lds double-buffered staging)
// ---------------------------------------------------------------------------

typedef __attribute__((ext_vector_type(16))) __bf16 v16bf;
typedef __attribute__((ext_vector_type(8)))  __bf16 v8bf;
typedef __attribute__((ext_vector_type(8)))  float  v8f;
typedef __attribute__((ext_vector_type(4)))  unsigned int u32x4;
typedef __attribute__((ext_vector_type(8)))  int i32x8;
typedef __attribute__((ext_vector_type(4)))  int i32x4;

constexpr int B_  = 4;
constexpr int C_  = 128;
constexpr int HW  = 64 * 64;     // N = 4096
constexpr int NT  = HW / 128;    // 32 tiles of 128
constexpr float LOG2E = 1.44269504088896340736f;

#if defined(__gfx1250__)
#if __has_builtin(__builtin_amdgcn_tensor_load_to_lds)
#define USE_TDM 1
#endif
#endif
#ifndef USE_TDM
#define USE_TDM 0
#endif

#if USE_TDM
// 2D TDM load: tile_d0 (contig elements, 2B each) x tile_d1 rows,
// row stride = stride_d0 elements. Descriptor per CDNA5 ISA 08_async_tensor §8.
__device__ __forceinline__ void tdm_load_2d(unsigned lds_addr, const void* gaddr,
                                            unsigned tile_d0, unsigned tile_d1,
                                            unsigned tensor_d0, unsigned tensor_d1,
                                            unsigned stride_d0) {
  unsigned long long ga = (unsigned long long)(uintptr_t)gaddr;
  u32x4 g0;
  g0[0] = 1u;                                        // count=1, user descriptor
  g0[1] = lds_addr;                                  // LDS byte address
  g0[2] = (unsigned)(ga & 0xffffffffu);              // global_addr[31:0]
  g0[3] = (unsigned)((ga >> 32) & 0x1ffffffu)        // global_addr[56:32]
        | (2u << 30);                                // type = 2 ("image")
  i32x8 g1;
  g1[0] = (int)(1u << 16);                           // data_size = 1 (2 bytes)
  g1[1] = (int)((tensor_d0 & 0xffffu) << 16);        // tensor_dim0[15:0]
  g1[2] = (int)(((tensor_d0 >> 16) & 0xffffu) | ((tensor_d1 & 0xffffu) << 16));
  g1[3] = (int)(((tensor_d1 >> 16) & 0xffffu) | ((tile_d0 & 0xffffu) << 16));
  g1[4] = (int)(tile_d1 & 0xffffu);                  // tile_dim1 (tile_dim2 = 0)
  g1[5] = (int)stride_d0;                            // tensor_dim0_stride[31:0]
  g1[6] = 0;
  g1[7] = 0;
  i32x4 z4 = {0, 0, 0, 0};
#if __has_include(<hip/amd_detail/amd_gfx1250_TDM.h>)
  i32x8 z8 = {0, 0, 0, 0, 0, 0, 0, 0};
  __builtin_amdgcn_tensor_load_to_lds(g0, g1, z4, z4, z8, 0);
#else
  __builtin_amdgcn_tensor_load_to_lds(g0, g1, z4, z4, 0);
#endif
}
#endif  // USE_TDM

// A/B bf16 fragment load per documented 16-bit 16x32 layout:
// lanes 0-15: elems 0..7 = K base..+7,    elems 8..15 = K base+16..+23
// lanes16-31: elems 0..7 = K base+8..+15, elems 8..15 = K base+24..+31
__device__ __forceinline__ v16bf load_frag(const __bf16* base, int row, int stride,
                                           int kbase, int half) {
  const __bf16* p = base + (size_t)row * stride + kbase + half * 8;
  v8bf lo = *(const v8bf*)(p);
  v8bf hi = *(const v8bf*)(p + 16);
  v16bf r;
#pragma unroll
  for (int i = 0; i < 8; ++i) { r[i] = lo[i]; r[i + 8] = hi[i]; }
  return r;
}

__device__ __forceinline__ v8f wmma_bf16(v16bf a, v16bf b, v8f c) {
  return __builtin_amdgcn_wmma_f32_16x16x32_bf16(false, a, false, b,
                                                 (short)0, c, false, false);
}

// ---------------------------------------------------------------------------
// Kernel 0: prep — xT (bf16, [B][N][C]), bf16 weights, wsum=wd1+wd2, bsum
// ---------------------------------------------------------------------------
__global__ void k_prep(const float* __restrict__ x,
                       const float* __restrict__ w1, const float* __restrict__ w2,
                       const float* __restrict__ w3,
                       const float* __restrict__ wd1, const float* __restrict__ wd2,
                       const float* __restrict__ bd1, const float* __restrict__ bd2,
                       __bf16* __restrict__ xT,
                       __bf16* __restrict__ w1b, __bf16* __restrict__ w2b,
                       __bf16* __restrict__ w3b, __bf16* __restrict__ wsb,
                       float* __restrict__ bsum) {
  int idx = blockIdx.x * blockDim.x + threadIdx.x;
  if (idx < B_ * HW * C_) {
    int c = idx % C_;
    int n = (idx / C_) % HW;
    int b = idx / (C_ * HW);
    xT[idx] = (__bf16)x[((size_t)b * C_ + c) * HW + n];
  }
  if (idx < C_ * C_) {
    w1b[idx] = (__bf16)w1[idx];
    w2b[idx] = (__bf16)w2[idx];
    w3b[idx] = (__bf16)w3[idx];
    wsb[idx] = (__bf16)(wd1[idx] + wd2[idx]);
  }
  if (idx < C_) bsum[idx] = bd1[idx] + bd2[idx];
}

// ---------------------------------------------------------------------------
// Kernel 1: projections  b_,c_,d_ = Wp @ x + bp   (WMMA, bf16 out)
//   bT,cT stored [B][N][C] (K=c contiguous);  dP stored [B][C][N]
// ---------------------------------------------------------------------------
__global__ void __launch_bounds__(256)
k_proj(const __bf16* __restrict__ xT,
       const __bf16* __restrict__ w1b, const __bf16* __restrict__ w2b,
       const __bf16* __restrict__ w3b,
       const float* __restrict__ b1, const float* __restrict__ b2,
       const float* __restrict__ b3,
       __bf16* __restrict__ bT, __bf16* __restrict__ cT, __bf16* __restrict__ dP) {
  const int wg   = blockIdx.x;
  const int b    = wg / NT;
  const int nt   = wg % NT;
  const int n0   = nt * 128;
  const int tid  = threadIdx.x;
  const int wave = tid >> 5;
  const int lane = tid & 31;
  const int half = lane >> 4;
  const int lm   = lane & 15;

  const __bf16* xTb = xT + (size_t)b * HW * C_;
  const __bf16* Ws[3] = {w1b, w2b, w3b};
  const float*  Bs[3] = {b1, b2, b3};

#pragma unroll
  for (int p = 0; p < 3; ++p) {
    v16bf fw[4];
#pragma unroll
    for (int kc = 0; kc < 4; ++kc)
      fw[kc] = load_frag(Ws[p], 16 * wave + lm, C_, kc * 32, half);
    for (int sub = 0; sub < 8; ++sub) {
      v8f acc = {};
#pragma unroll
      for (int kc = 0; kc < 4; ++kc) {
        v16bf fx = load_frag(xTb, n0 + 16 * sub + lm, C_, kc * 32, half);
        acc = wmma_bf16(fw[kc], fx, acc);
      }
#pragma unroll
      for (int r = 0; r < 8; ++r) {
        int o = 16 * wave + r + 8 * half;   // output channel (row M)
        int n = n0 + 16 * sub + lm;         // pixel (col N)
        __bf16 v = (__bf16)(acc[r] + Bs[p][o]);
        if (p == 0)      bT[((size_t)b * HW + n) * C_ + o] = v;
        else if (p == 1) cT[((size_t)b * HW + n) * C_ + o] = v;
        else             dP[((size_t)b * C_ + o) * HW + n] = v;
      }
    }
  }
}

// ---------------------------------------------------------------------------
// Kernel 2: column-softmax stats. scores[i,j] = sum_c c_[c,i]*b_[c,j]
//   m[b,j] = max_i (scores*LOG2E);  Z[b,j] = sum_i exp2(scaled - m)
//   wave w owns j columns [jt*128 + 16w .. +15], loops all i.
//   cT i-rows (shared by all 8 waves) staged into LDS by TDM, double-buffered.
// ---------------------------------------------------------------------------
__global__ void __launch_bounds__(256)
k_stats(const __bf16* __restrict__ cT, const __bf16* __restrict__ bT,
        float* __restrict__ m_out, float* __restrict__ z_out) {
  constexpr int ROWS = 64;            // i-rows per stage
  constexpr int NSTG = HW / ROWS;     // 64 stages
#if USE_TDM
  __shared__ __bf16 stage[2 * ROWS * C_];   // 32 KB double buffer
#endif
  const int wg   = blockIdx.x;
  const int b    = wg / NT;
  const int jt   = wg % NT;
  const int tid  = threadIdx.x;
  const int wave = tid >> 5;
  const int lane = tid & 31;
  const int half = lane >> 4;
  const int lm   = lane & 15;
  const int j0   = jt * 128 + wave * 16;

  const __bf16* cTb = cT + (size_t)b * HW * C_;
  const __bf16* bTb = bT + (size_t)b * HW * C_;

  v16bf fb[4];
#pragma unroll
  for (int kc = 0; kc < 4; ++kc)
    fb[kc] = load_frag(bTb, j0 + lm, C_, kc * 32, half);   // fixed B (this wave's j)

  float m = -1e30f, s = 0.0f;

#if USE_TDM
  if (wave == 0)
    tdm_load_2d((unsigned)(uintptr_t)stage, cTb, C_, ROWS, C_, HW, C_);
#endif
  for (int st = 0; st < NSTG; ++st) {
#if USE_TDM
    const __bf16* abase = stage + (st & 1) * (ROWS * C_);
    __builtin_amdgcn_s_wait_tensorcnt(0);
    __syncthreads();
    if (wave == 0 && st + 1 < NSTG)
      tdm_load_2d((unsigned)(uintptr_t)(stage + ((st + 1) & 1) * (ROWS * C_)),
                  cTb + (size_t)(st + 1) * ROWS * C_, C_, ROWS, C_, HW, C_);
#endif
#pragma unroll
    for (int it = 0; it < ROWS / 16; ++it) {
      v8f acc = {};
#pragma unroll
      for (int kc = 0; kc < 4; ++kc) {
#if USE_TDM
        v16bf fa = load_frag(abase, it * 16 + lm, C_, kc * 32, half);
#else
        v16bf fa = load_frag(cTb, st * ROWS + it * 16 + lm, C_, kc * 32, half);
#endif
        acc = wmma_bf16(fa, fb[kc], acc);
      }
#pragma unroll
      for (int r = 0; r < 8; ++r) {         // 8 rows (i) of this lane's column j
        float v  = acc[r] * LOG2E;
        float mn = fmaxf(m, v);
        s = s * exp2f(m - mn) + exp2f(v - mn);
        m = mn;
      }
    }
  }
  // lanes l and l^16 hold the two row-halves of the same column j0+lm
  float mo = __shfl_xor(m, 16, 32);
  float so = __shfl_xor(s, 16, 32);
  float mn = fmaxf(m, mo);
  float zt = s * exp2f(m - mn) + so * exp2f(mo - mn);
  if (half == 0) {
    m_out[(size_t)b * HW + j0 + lm] = mn;
    z_out[(size_t)b * HW + j0 + lm] = zt;
  }
}

// ---------------------------------------------------------------------------
// Kernel 3: attn[c,k] = sum_j d[c,j]*P[k,j], P[k,j]=exp2(s*LOG2E - m_j)/Z_j
//   then e = x + alpha*attn, then out = (wd1+wd2) @ e + (bd1+bd2)  (fused)
//   One workgroup = one (batch, 128-wide k-tile). 8 waves:
//     scores phase: wave w computes score rows k0+16w..+15 for the j-stage.
//     attn phase:   wave w owns channel rows 16w..+15, all 128 k columns.
//   bT j-tile (shared by all waves) staged by TDM, double-buffered.
// ---------------------------------------------------------------------------
__global__ void __launch_bounds__(256)
k_attn(const __bf16* __restrict__ cT, const __bf16* __restrict__ bT,
       const __bf16* __restrict__ dP,
       const float* __restrict__ m_in, const float* __restrict__ z_in,
       const float* __restrict__ x, const float* __restrict__ alpha_p,
       const __bf16* __restrict__ wsb, const float* __restrict__ bsum,
       float* __restrict__ out) {
  constexpr int JT   = 32;            // j columns per stage
  constexpr int NSTG = HW / JT;       // 128 stages
  __shared__ __bf16 P_lds[128 * JT];    // P tile [k][j], 8 KB
  __shared__ __bf16 e_lds[128 * 128];   // e tile [k][c], 32 KB (epilogue)
#if USE_TDM
  __shared__ __bf16 bstage[2 * JT * C_];  // 16 KB double buffer for bT tile
#endif

  const int wg   = blockIdx.x;
  const int b    = wg / NT;
  const int kt   = wg % NT;
  const int k0   = kt * 128;
  const int tid  = threadIdx.x;
  const int wave = tid >> 5;
  const int lane = tid & 31;
  const int half = lane >> 4;
  const int lm   = lane & 15;

  const __bf16* cTb = cT + (size_t)b * HW * C_;
  const __bf16* bTb = bT + (size_t)b * HW * C_;
  const __bf16* dPb = dP + (size_t)b * C_ * HW;
  const float*  mB  = m_in + (size_t)b * HW;
  const float*  zB  = z_in + (size_t)b * HW;

  // scores A fragments: this wave's k rows (fixed for whole kernel)
  v16bf faS[4];
#pragma unroll
  for (int kc = 0; kc < 4; ++kc)
    faS[kc] = load_frag(cTb, k0 + 16 * wave + lm, C_, kc * 32, half);

  v8f accA[8];
#pragma unroll
  for (int n = 0; n < 8; ++n) accA[n] = {};

#if USE_TDM
  if (wave == 0)
    tdm_load_2d((unsigned)(uintptr_t)bstage, bTb, C_, JT, C_, HW, C_);
#endif

  for (int st = 0; st < NSTG; ++st) {
    const int j0s = st * JT;
#if USE_TDM
    const __bf16* bbuf = bstage + (st & 1) * (JT * C_);
    __builtin_amdgcn_s_wait_tensorcnt(0);
#endif
    __syncthreads();   // bT tile ready; prev P_lds fully consumed
#if USE_TDM
    if (wave == 0 && st + 1 < NSTG)
      tdm_load_2d((unsigned)(uintptr_t)(bstage + ((st + 1) & 1) * (JT * C_)),
                  bTb + (size_t)(j0s + JT) * C_, C_, JT, C_, HW, C_);
#endif
    // ---- scores -> P tile (each wave writes its own 16 k rows) ----
#pragma unroll
    for (int sub = 0; sub < 2; ++sub) {
      v8f sc = {};
#pragma unroll
      for (int kc = 0; kc < 4; ++kc) {
#if USE_TDM
        v16bf fbB = load_frag(bbuf, 16 * sub + lm, C_, kc * 32, half);
#else
        v16bf fbB = load_frag(bTb, j0s + 16 * sub + lm, C_, kc * 32, half);
#endif
        sc = wmma_bf16(faS[kc], fbB, sc);
      }
      int j    = j0s + 16 * sub + lm;     // this lane's column
      float mj = mB[j];
      float rz = 1.0f / zB[j];
#pragma unroll
      for (int r = 0; r < 8; ++r) {
        float p = exp2f(sc[r] * LOG2E - mj) * rz;
        int krow = 16 * wave + r + 8 * half;
        P_lds[krow * JT + 16 * sub + lm] = (__bf16)p;
      }
    }
    __syncthreads();
    // ---- attn += d[:, jtile] @ P^T ----
    v16bf fad = load_frag(dPb + (size_t)(16 * wave + lm) * HW, 0, 0, j0s, half);
#pragma unroll
    for (int n = 0; n < 8; ++n) {
      v16bf fbP = load_frag(P_lds, 16 * n + lm, JT, 0, half);
      accA[n] = wmma_bf16(fad, fbP, accA[n]);
    }
  }
  __syncthreads();

  // ---- epilogue: e = x + alpha*attn -> e_lds[k][c] (bf16) ----
  const float alpha = *alpha_p;
#pragma unroll
  for (int n = 0; n < 8; ++n) {
#pragma unroll
    for (int r = 0; r < 8; ++r) {
      int c = 16 * wave + r + 8 * half;
      int k = 16 * n + lm;
      float e = x[((size_t)b * C_ + c) * HW + k0 + k] + alpha * accA[n][r];
      e_lds[k * C_ + c] = (__bf16)e;
    }
  }
  __syncthreads();

  // ---- out = wsum @ e + bsum  (wave w: output rows 16w..+15) ----
  v16bf faW[4];
#pragma unroll
  for (int kc = 0; kc < 4; ++kc)
    faW[kc] = load_frag(wsb, 16 * wave + lm, C_, kc * 32, half);
#pragma unroll
  for (int n = 0; n < 8; ++n) {
    v8f o = {};
#pragma unroll
    for (int kc = 0; kc < 4; ++kc) {
      v16bf fbE = load_frag(e_lds, 16 * n + lm, C_, kc * 32, half);
      o = wmma_bf16(faW[kc], fbE, o);
    }
#pragma unroll
    for (int r = 0; r < 8; ++r) {
      int oc = 16 * wave + r + 8 * half;
      out[((size_t)b * C_ + oc) * HW + k0 + 16 * n + lm] = o[r] + bsum[oc];
    }
  }
}

// ---------------------------------------------------------------------------
// Launch
// ---------------------------------------------------------------------------
extern "C" void kernel_launch(void* const* d_in, const int* in_sizes, int n_in,
                              void* d_out, int out_size, void* d_ws, size_t ws_size,
                              hipStream_t stream) {
  const float* x     = (const float*)d_in[0];
  const float* w1    = (const float*)d_in[1];
  const float* b1    = (const float*)d_in[2];
  const float* w2    = (const float*)d_in[3];
  const float* b2    = (const float*)d_in[4];
  const float* w3    = (const float*)d_in[5];
  const float* b3    = (const float*)d_in[6];
  const float* wd1   = (const float*)d_in[7];
  const float* bd1   = (const float*)d_in[8];
  const float* wd2   = (const float*)d_in[9];
  const float* bd2   = (const float*)d_in[10];
  const float* alpha = (const float*)d_in[11];
  float* out = (float*)d_out;
  (void)b2; (void)ws_size; (void)in_sizes; (void)n_in; (void)out_size;

  char* ws = (char*)d_ws;
  size_t off = 0;
  auto alloc = [&](size_t bytes) { char* p = ws + off; off = (off + bytes + 255) & ~size_t(255); return p; };

  __bf16* xT   = (__bf16*)alloc((size_t)B_ * HW * C_ * 2);   // 4 MB
  __bf16* bT   = (__bf16*)alloc((size_t)B_ * HW * C_ * 2);   // 4 MB
  __bf16* cTm  = (__bf16*)alloc((size_t)B_ * HW * C_ * 2);   // 4 MB
  __bf16* dPm  = (__bf16*)alloc((size_t)B_ * HW * C_ * 2);   // 4 MB
  __bf16* w1b  = (__bf16*)alloc((size_t)C_ * C_ * 2);
  __bf16* w2b  = (__bf16*)alloc((size_t)C_ * C_ * 2);
  __bf16* w3b  = (__bf16*)alloc((size_t)C_ * C_ * 2);
  __bf16* wsb  = (__bf16*)alloc((size_t)C_ * C_ * 2);
  float*  bsum = (float*)alloc((size_t)C_ * 4);
  float*  mArr = (float*)alloc((size_t)B_ * HW * 4);
  float*  zArr = (float*)alloc((size_t)B_ * HW * 4);

  {
    int total = B_ * HW * C_;
    k_prep<<<(total + 255) / 256, 256, 0, stream>>>(x, w1, w2, w3, wd1, wd2, bd1, bd2,
                                                    xT, w1b, w2b, w3b, wsb, bsum);
  }
  k_proj<<<B_ * NT, 256, 0, stream>>>(xT, w1b, w2b, w3b, b1, b2, b3, bT, cTm, dPm);
  k_stats<<<B_ * NT, 256, 0, stream>>>(cTm, bT, mArr, zArr);
  k_attn<<<B_ * NT, 256, 0, stream>>>(cTm, bT, dPm, mArr, zArr, x, alpha, wsb, bsum, out);
}
